// Attention_87479893885682
// MI455X (gfx1250) — matrix operand order repaired and checked by reading, not compiled, not run on hardware
//
#include <hip/hip_runtime.h>

#define BB  4
#define SS  2048
#define DD  1024
#define DKK 128

typedef __bf16 bf16_t;
typedef __attribute__((ext_vector_type(16))) __bf16 bf16x16;
typedef __attribute__((ext_vector_type(8)))  __bf16 bf16x8;
typedef __attribute__((ext_vector_type(8)))  float  f32x8;

static __device__ __forceinline__ f32x8 wmma_bf16(bf16x16 a, bf16x16 b, f32x8 c) {
  // D = A(16x32 bf16) * B(32x16 bf16) + C(16x16 f32)
  return __builtin_amdgcn_wmma_f32_16x16x32_bf16(false, a, false, b, (short)0, c,
                                                 false, false);
}

static __device__ __forceinline__ bf16x16 join8(bf16x8 lo, bf16x8 hi) {
  bf16x16 r;
#pragma unroll
  for (int i = 0; i < 8; ++i) { r[i] = lo[i]; r[i + 8] = hi[i]; }
  return r;
}

static __device__ __forceinline__ f32x8 zero8() {
  f32x8 z = {0.f, 0.f, 0.f, 0.f, 0.f, 0.f, 0.f, 0.f};
  return z;
}

// ---------------- fp32 -> bf16 bulk convert ----------------
__global__ void k_f32_to_bf16(const float* __restrict__ src,
                              bf16_t* __restrict__ dst, int n) {
  int i = (blockIdx.x * blockDim.x + threadIdx.x) * 4;
  if (i >= n) return;
  float4 v = *(const float4*)(src + i);
  dst[i + 0] = (bf16_t)v.x;
  dst[i + 1] = (bf16_t)v.y;
  dst[i + 2] = (bf16_t)v.z;
  dst[i + 3] = (bf16_t)v.w;
}

// ---------------- weight transpose: Wt[n][k] = bf16(W[k][n]) ----------------
__global__ void k_transpose_w(const float* __restrict__ W,
                              bf16_t* __restrict__ Wt, int K, int N) {
  int t = blockIdx.x * blockDim.x + threadIdx.x;
  if (t >= K * N) return;
  int n = t / K, k = t - n * K;
  Wt[t] = (bf16_t)W[(size_t)k * N + n];
}

// ---------------- projection GEMM: Y = X @ W + b (bf16 WMMA) ----------------
// A: [M][Kd] bf16 row-major, Bt: [N][Kd] bf16 (W transposed).
// Each wave computes a 16(M) x 128(N) tile; 8 f32 16x16 accumulators.
// transposed=1 writes V^T layout [BB][DD][SS] instead of [M][N].
__global__ __launch_bounds__(256)
void k_gemm_bf16(const bf16_t* __restrict__ A, const bf16_t* __restrict__ Bt,
                 const float* __restrict__ bias, bf16_t* __restrict__ out,
                 int M, int N, int Kd, int nchunks, int transposed) {
  int w    = (int)((blockIdx.x * blockDim.x + threadIdx.x) >> 5);
  int lane = threadIdx.x & 31;
  int half = lane >> 4, l16 = lane & 15;
  int nc = w % nchunks;
  int mt = w / nchunks;
  if (mt * 16 >= M) return;  // wave-uniform
  int mbase = mt * 16, nbase = nc * 128;

  f32x8 acc[8];
#pragma unroll
  for (int i = 0; i < 8; ++i) acc[i] = zero8();

  const bf16_t* arow = A + (size_t)(mbase + l16) * Kd;
  for (int kc = 0; kc < Kd; kc += 32) {
    bf16x8 alo = *(const bf16x8*)(arow + kc + half * 16);
    bf16x8 ahi = *(const bf16x8*)(arow + kc + half * 16 + 8);
    bf16x16 af = join8(alo, ahi);
#pragma unroll
    for (int nt = 0; nt < 8; ++nt) {
      const bf16_t* bp = Bt + (size_t)(nbase + nt * 16 + l16) * Kd + kc + half * 16;
      acc[nt] = wmma_bf16(af, *(const bf16x16*)bp, acc[nt]);
    }
  }

#pragma unroll
  for (int nt = 0; nt < 8; ++nt) {
    int n = nbase + nt * 16 + l16;
    float bv = bias[n];
#pragma unroll
    for (int r = 0; r < 8; ++r) {
      int m = mbase + (half ? 8 + r : r);
      float v = acc[nt][r] + bv;
      if (!transposed) {
        out[(size_t)m * N + n] = (bf16_t)v;
      } else {  // V^T: [BB][DD][SS]
        int bb = m >> 11;         // m / SS
        int s  = m & (SS - 1);    // m % SS
        out[((size_t)bb * DD + n) * SS + s] = (bf16_t)v;
      }
    }
  }
}

// ---------------- per-batch column mean of V (for fully-padded rows) --------
__global__ void k_mean_v(const bf16_t* __restrict__ VT, float* __restrict__ meanv) {
  int idx = blockIdx.x * blockDim.x + threadIdx.x;  // over BB*DD
  if (idx >= BB * DD) return;
  const bf16_t* p = VT + (size_t)idx * SS;
  float s = 0.f;
  for (int i = 0; i < SS; i += 8) {
    bf16x8 v = *(const bf16x8*)(p + i);
#pragma unroll
    for (int j = 0; j < 8; ++j) s += (float)v[j];
  }
  meanv[idx] = s * (1.0f / SS);
}

// ---------------- flash attention (causal only; pad rows fixed up later) ----
// wave w -> (b, query-tile qt of 16, D-chunk of 128). 32-key tiles, online
// softmax entirely in registers; P relayout C->A via per-wave LDS staging.
__global__ __launch_bounds__(256)
void k_flash_attn(const bf16_t* __restrict__ Q, const bf16_t* __restrict__ Kb,
                  const bf16_t* __restrict__ VT, float* __restrict__ out) {
  __shared__ __align__(16) bf16_t psh[8][16 * 32];
  int wv   = threadIdx.x >> 5;
  int lane = threadIdx.x & 31;
  int half = lane >> 4, l16 = lane & 15;
  int w      = blockIdx.x * 8 + wv;
  int dchunk = w & 7;
  int qt     = (w >> 3) & 127;
  int b      = w >> 10;

  // Q fragments for 16 rows x full DK=128 (4 chunks of 32)
  bf16x16 qf[4];
  const bf16_t* qrow = Q + ((size_t)b * SS + qt * 16 + l16) * DKK;
#pragma unroll
  for (int c = 0; c < 4; ++c) {
    bf16x8 lo = *(const bf16x8*)(qrow + c * 32 + half * 16);
    bf16x8 hi = *(const bf16x8*)(qrow + c * 32 + half * 16 + 8);
    qf[c] = join8(lo, hi);
  }

  f32x8 o[8];
#pragma unroll
  for (int i = 0; i < 8; ++i) o[i] = zero8();
  float mrow[8], lrow[8];
#pragma unroll
  for (int r = 0; r < 8; ++r) { mrow[r] = -1e30f; lrow[r] = 0.f; }

  const float scale = 0.08838834764831845f;  // 1/sqrt(128)
  int nkt = (((qt + 1) * 16) + 31) >> 5;

  for (int kt = 0; kt < nkt; ++kt) {
    int kb = kt * 32;
    const bf16_t* kp0 = Kb + ((size_t)b * SS + kb + l16) * DKK + half * 16;
    const bf16_t* kp1 = kp0 + 16 * DKK;
    const bf16_t* vp  = VT + ((size_t)b * DD + dchunk * 128 + l16) * SS + kb + half * 16;
    if (kt + 1 < nkt) {  // prefetch next key tile (global_prefetch_b8)
      __builtin_prefetch(kp0 + 32 * DKK, 0, 0);
      __builtin_prefetch(vp + 32, 0, 0);
    }

    // scores: two 16x16 tiles (key halves), chained over 4 dk-chunks
    f32x8 s0 = zero8(), s1 = zero8();
#pragma unroll
    for (int c = 0; c < 4; ++c) {
      s0 = wmma_bf16(qf[c], *(const bf16x16*)(kp0 + c * 32), s0);
      s1 = wmma_bf16(qf[c], *(const bf16x16*)(kp1 + c * 32), s1);
    }

    int key0 = kb + l16, key1 = kb + 16 + l16;
    float corr[8];
#pragma unroll
    for (int r = 0; r < 8; ++r) {
      int q  = qt * 16 + (half ? 8 + r : r);
      float a0 = (key0 <= q) ? s0[r] * scale : -1e30f;
      float a1 = (key1 <= q) ? s1[r] * scale : -1e30f;
      // row max across 32 key columns (xor stays within 16-lane half)
      float mx = fmaxf(a0, a1);
      mx = fmaxf(mx, __shfl_xor(mx, 1));
      mx = fmaxf(mx, __shfl_xor(mx, 2));
      mx = fmaxf(mx, __shfl_xor(mx, 4));
      mx = fmaxf(mx, __shfl_xor(mx, 8));
      float mnew = fmaxf(mrow[r], mx);
      float cr   = __expf(mrow[r] - mnew);
      float p0   = __expf(a0 - mnew);
      float p1   = __expf(a1 - mnew);
      float rs = p0 + p1;
      rs += __shfl_xor(rs, 1);
      rs += __shfl_xor(rs, 2);
      rs += __shfl_xor(rs, 4);
      rs += __shfl_xor(rs, 8);
      lrow[r] = lrow[r] * cr + rs;
      mrow[r] = mnew;
      corr[r] = cr;
      int row = half ? 8 + r : r;
      psh[wv][row * 32 + l16]      = (bf16_t)p0;
      psh[wv][row * 32 + 16 + l16] = (bf16_t)p1;
    }

    // rescale running output by per-row correction
#pragma unroll
    for (int nt = 0; nt < 8; ++nt)
#pragma unroll
      for (int r = 0; r < 8; ++r) o[nt][r] *= corr[r];

    // reload P as an A fragment (16x32) from LDS (same-wave, DS in-order)
    bf16x8 plo = *(const bf16x8*)(&psh[wv][l16 * 32 + half * 16]);
    bf16x8 phi = *(const bf16x8*)(&psh[wv][l16 * 32 + half * 16 + 8]);
    bf16x16 pf = join8(plo, phi);

    // O += P @ V over this wave's 128 output columns
#pragma unroll
    for (int nt = 0; nt < 8; ++nt)
      o[nt] = wmma_bf16(pf, *(const bf16x16*)(vp + (size_t)nt * 16 * SS), o[nt]);
  }

  float inv[8];
#pragma unroll
  for (int r = 0; r < 8; ++r) inv[r] = 1.0f / fmaxf(lrow[r], 1e-30f);

  float* ob = out + ((size_t)b * SS + qt * 16) * DD + dchunk * 128 + l16;
#pragma unroll
  for (int nt = 0; nt < 8; ++nt)
#pragma unroll
    for (int r = 0; r < 8; ++r) {
      int m = half ? 8 + r : r;
      ob[(size_t)m * DD + nt * 16] = o[nt][r] * inv[r];
    }
}

// ---------------- fully-padded query rows -> uniform mean of V --------------
__global__ void k_fixup(const unsigned char* __restrict__ mask,
                        const float* __restrict__ meanv,
                        float* __restrict__ out) {
  int bq = blockIdx.x;
  if (!mask[bq]) return;
  int b = bq >> 11;  // / SS
  const float* mv = meanv + (size_t)b * DD;
  float* op = out + (size_t)bq * DD;
  for (int d = threadIdx.x; d < DD; d += blockDim.x) op[d] = mv[d];
}

extern "C" void kernel_launch(void* const* d_in, const int* in_sizes, int n_in,
                              void* d_out, int out_size, void* d_ws, size_t ws_size,
                              hipStream_t stream) {
  (void)in_sizes; (void)n_in; (void)out_size; (void)ws_size;
  const float* x            = (const float*)d_in[0];
  const unsigned char* pmsk = (const unsigned char*)d_in[1];
  const float* Wq = (const float*)d_in[2];
  const float* bq = (const float*)d_in[3];
  const float* Wk = (const float*)d_in[4];
  const float* bk = (const float*)d_in[5];
  const float* Wv = (const float*)d_in[6];
  const float* bv = (const float*)d_in[7];
  float* out = (float*)d_out;

  char* ws = (char*)d_ws;
  size_t off = 0;
  auto take = [&](size_t bytes) -> char* {
    char* p = ws + off;
    off += (bytes + 255) & ~(size_t)255;
    return p;
  };
  bf16_t* Xbf   = (bf16_t*)take((size_t)BB * SS * DD * 2);   // 16 MB
  bf16_t* WqT   = (bf16_t*)take((size_t)DD * DKK * 2);
  bf16_t* WkT   = (bf16_t*)take((size_t)DD * DKK * 2);
  bf16_t* WvT   = (bf16_t*)take((size_t)DD * DD * 2);
  bf16_t* Qbf   = (bf16_t*)take((size_t)BB * SS * DKK * 2);
  bf16_t* Kbf   = (bf16_t*)take((size_t)BB * SS * DKK * 2);
  bf16_t* VT    = (bf16_t*)take((size_t)BB * DD * SS * 2);   // 16 MB
  float*  meanv = (float*)take((size_t)BB * DD * 4);

  const int nX = BB * SS * DD;
  k_f32_to_bf16<<<(nX / 4 + 255) / 256, 256, 0, stream>>>(x, Xbf, nX);
  k_transpose_w<<<(DD * DKK + 255) / 256, 256, 0, stream>>>(Wq, WqT, DD, DKK);
  k_transpose_w<<<(DD * DKK + 255) / 256, 256, 0, stream>>>(Wk, WkT, DD, DKK);
  k_transpose_w<<<(DD * DD + 255) / 256, 256, 0, stream>>>(Wv, WvT, DD, DD);

  // Q,K: 512 m-tiles x 1 n-chunk = 512 waves = 64 blocks
  k_gemm_bf16<<<64, 256, 0, stream>>>(Xbf, WqT, bq, Qbf, BB * SS, DKK, DD, 1, 0);
  k_gemm_bf16<<<64, 256, 0, stream>>>(Xbf, WkT, bk, Kbf, BB * SS, DKK, DD, 1, 0);
  // V (transposed out): 512 x 8 = 4096 waves = 512 blocks
  k_gemm_bf16<<<512, 256, 0, stream>>>(Xbf, WvT, bv, VT, BB * SS, DD, DD, 8, 1);

  k_mean_v<<<(BB * DD + 255) / 256, 256, 0, stream>>>(VT, meanv);

  // 4 batches * 128 query tiles * 8 d-chunks = 4096 waves = 512 blocks
  k_flash_attn<<<512, 256, 0, stream>>>(Qbf, Kbf, VT, out);

  k_fixup<<<BB * SS, 256, 0, stream>>>(pmsk, meanv, out);
}
